// TopExtraAttention_48679159333610
// MI455X (gfx1250) — matrix operand order repaired and checked by reading, not compiled, not run on hardware
//
#include <hip/hip_runtime.h>

typedef __attribute__((ext_vector_type(16))) _Float16 v16h;
typedef __attribute__((ext_vector_type(8)))  _Float16 h8;
typedef __attribute__((ext_vector_type(8)))  float    v8f;

typedef int v4i __attribute__((vector_size(16)));
typedef __attribute__((address_space(1))) v4i as1_v4i;
typedef __attribute__((address_space(3))) v4i as3_v4i;

#if __has_builtin(__builtin_amdgcn_global_load_async_to_lds_b128) && \
    __has_builtin(__builtin_amdgcn_s_wait_asynccnt)
#define USE_ASYNC 1
#else
#define USE_ASYNC 0
#endif

#define B_   8
#define SQ_  2048
#define SK_  4096
#define D_   512
#define DV_  256
#define KT   32          // keys per tile
#define QW   16          // query rows per wave
#define WAVES 4
#define QB   (QW*WAVES)  // 64 query rows per block

__device__ __forceinline__ v8f wmma_f16(v16h a, v16h b, v8f c) {
  // D = A(16x32 f16) * B(32x16 f16) + C(16x16 f32)
  return __builtin_amdgcn_wmma_f32_16x16x32_f16(false, a, false, b, (short)0, c,
                                                false, false);
}

__device__ __forceinline__ void put4(v16h& v, int i, float4 f) {
  v[i]   = (_Float16)f.x; v[i+1] = (_Float16)f.y;
  v[i+2] = (_Float16)f.z; v[i+3] = (_Float16)f.w;
}

// Load a B-fragment: 16 contiguous f16 (K dimension) at p for this lane.
__device__ __forceinline__ v16h ldb(const _Float16* p) {
  h8 lo = *(const h8*)p;
  h8 hi = *(const h8*)(p + 8);
  v16h r;
#pragma unroll
  for (int i = 0; i < 8; ++i) { r[i] = lo[i]; r[8+i] = hi[i]; }
  return r;
}

// 16-byte global -> LDS copy, async when available.
__device__ __forceinline__ void cp16(const _Float16* g, _Float16* l) {
#if USE_ASYNC
  __builtin_amdgcn_global_load_async_to_lds_b128(
      (as1_v4i*)(unsigned long long)g,
      (as3_v4i*)(unsigned)(unsigned long long)l, 0, 0);
#else
  *(h8*)l = *(const h8*)g;
#endif
}

// ---------------------------------------------------------------------------
// Projection GEMM: C[m][n] = dot(P[m][0..511], Q[n][0..511]), stored as
// Out[n*ldo + m] in f16. One 16x16 tile per wave, K swept with 16 WMMAs.
// ---------------------------------------------------------------------------
__global__ __launch_bounds__(128) void proj_gemm16(
    const float* __restrict__ P, const float* __restrict__ Q,
    _Float16* __restrict__ Out, int nTiles, int totalTiles, int ldo)
{
  const int w = blockIdx.x * 4 + (threadIdx.x >> 5);
  if (w >= totalTiles) return;
  const int lane = threadIdx.x & 31;
  const int hs   = lane >> 4;
  const int lr   = lane & 15;
  const int mt   = w / nTiles, nt = w % nTiles;

  const float* Pr = P + (size_t)(mt*16 + lr) * D_;
  const float* Qr = Q + (size_t)(nt*16 + lr) * D_;

  v8f acc = {};
#pragma unroll
  for (int kk = 0; kk < 16; ++kk) {
    const int k0 = kk * 32;
    v16h a;
    put4(a, 0,  *(const float4*)(Pr + k0 + hs*8));
    put4(a, 4,  *(const float4*)(Pr + k0 + hs*8 + 4));
    put4(a, 8,  *(const float4*)(Pr + k0 + 16 + hs*8));
    put4(a, 12, *(const float4*)(Pr + k0 + 16 + hs*8 + 4));
    v16h b;
    put4(b, 0,  *(const float4*)(Qr + k0 + hs*8));
    put4(b, 4,  *(const float4*)(Qr + k0 + hs*8 + 4));
    put4(b, 8,  *(const float4*)(Qr + k0 + 16 + hs*8));
    put4(b, 12, *(const float4*)(Qr + k0 + 16 + hs*8 + 4));
    acc = wmma_f16(a, b, acc);
  }
  h8 o;
#pragma unroll
  for (int r = 0; r < 8; ++r) o[r] = (_Float16)acc[r];
  *(h8*)(Out + (size_t)(nt*16 + lr) * ldo + mt*16 + hs*8) = o;
}

// ---------------------------------------------------------------------------
// Flash attention. Kh: [SK][D] f16, Vt: [DV][SK] f16 (transposed).
// Block = 4 waves, 64 query rows; wave owns 16 rows, full DV in registers;
// 32-key tiles double-buffered in LDS via async-to-LDS when available.
// ---------------------------------------------------------------------------
__global__ __launch_bounds__(WAVES*32) void attn_flash(
    const float* __restrict__ X, const _Float16* __restrict__ Kh,
    const _Float16* __restrict__ Vt, float* __restrict__ Out)
{
#if USE_ASYNC
  __shared__ __align__(16) _Float16 Ksh[2][KT][D_ + 8];        // 2 x 32 x 520
  __shared__ __align__(16) _Float16 Vsh[2][DV_][KT + 8];       // 2 x 256 x 40
#else
  __shared__ __align__(16) _Float16 Ksh[1][KT][D_ + 8];
  __shared__ __align__(16) _Float16 Vsh[1][DV_][KT + 8];
#endif
  __shared__ __align__(16) _Float16 Psh[WAVES][16][KT + 8];    // per-wave P

  const int tid  = threadIdx.x;
  const int wave = tid >> 5;
  const int lane = tid & 31;
  const int hs   = lane >> 4;
  const int lr   = lane & 15;
  const int b    = blockIdx.y;
  const int q0   = blockIdx.x * QB + wave * QW;
  const float scale = 0.044194173824159216f;  // 1/sqrt(512)
  const int NT = SK_ / KT;

  // stage one 32-key tile (K: 16 chunks/thread, V: 8 chunks/thread)
  auto stage = [&](int buf, int kt) {
    for (int i = tid; i < KT * (D_/8); i += WAVES*32) {
      const int row = i >> 6;            // D_/8 == 64 chunks per row
      const int c   = (i & 63) * 8;
      cp16(&Kh[(size_t)(kt*KT + row)*D_ + c], &Ksh[buf][row][c]);
    }
    for (int i = tid; i < DV_ * (KT/8); i += WAVES*32) {
      const int row = i >> 2;
      const int c   = (i & 3) * 8;
      cp16(&Vt[(size_t)row*SK_ + kt*KT + c], &Vsh[buf][row][c]);
    }
  };

  // ---- Q fragments (f32 -> f16), kept in registers: 16 x v16h
  v16h qa[16];
  {
    const float* Xr = X + ((size_t)b * SQ_ + q0 + lr) * D_;
#pragma unroll
    for (int kk = 0; kk < 16; ++kk) {
      const int k0 = kk*32 + hs*16;
      put4(qa[kk], 0,  *(const float4*)(Xr + k0));
      put4(qa[kk], 4,  *(const float4*)(Xr + k0 + 4));
      put4(qa[kk], 8,  *(const float4*)(Xr + k0 + 8));
      put4(qa[kk], 12, *(const float4*)(Xr + k0 + 12));
    }
  }

  v8f acc[16];
  const v8f vzero = {};
#pragma unroll
  for (int j = 0; j < 16; ++j) acc[j] = vzero;
  float rmax[8], rsum[8];
#pragma unroll
  for (int r = 0; r < 8; ++r) { rmax[r] = -3.0e38f; rsum[r] = 0.0f; }

#if USE_ASYNC
  stage(0, 0);                           // preload first tile
#endif

  for (int kt = 0; kt < NT; ++kt) {
#if USE_ASYNC
    const int cur = kt & 1;
    __syncthreads();                     // all waves done reading buf cur^1
    if (kt + 1 < NT) {
      stage(cur ^ 1, kt + 1);            // kick off next tile (24 ops/thread)
      __builtin_amdgcn_s_wait_asynccnt(24);   // current tile resident
    } else {
      __builtin_amdgcn_s_wait_asynccnt(0);
    }
    __syncthreads();                     // every wave's async writes visible
#else
    const int cur = 0;
    __syncthreads();
    stage(0, kt);
    if (kt + 1 < NT) {
      __builtin_prefetch(&Kh[(size_t)(kt+1)*KT*D_ + (size_t)tid*16], 0, 3);
      __builtin_prefetch(&Vt[(size_t)(tid*2)*SK_ + (kt+1)*KT], 0, 3);
    }
    __syncthreads();
#endif

    // ---- S = Q * K^T : two 16x16 f32 tiles (keys 0..15 and 16..31)
    v8f s0 = vzero, s1 = vzero;
#pragma unroll
    for (int kk = 0; kk < 16; ++kk) {
      const int c = kk*32 + hs*16;
      v16h b0 = ldb(&Ksh[cur][lr][c]);
      v16h b1 = ldb(&Ksh[cur][16 + lr][c]);
      s0 = wmma_f16(qa[kk], b0, s0);
      s1 = wmma_f16(qa[kk], b1, s1);
    }

    // ---- online softmax (each row lives in one 16-lane group)
#pragma unroll
    for (int r = 0; r < 8; ++r) {
      const float a0 = s0[r] * scale, a1 = s1[r] * scale;
      float t = fmaxf(a0, a1);
      t = fmaxf(t, __shfl_xor(t, 1, 16));
      t = fmaxf(t, __shfl_xor(t, 2, 16));
      t = fmaxf(t, __shfl_xor(t, 4, 16));
      t = fmaxf(t, __shfl_xor(t, 8, 16));
      const float nm    = fmaxf(rmax[r], t);
      const float alpha = __expf(rmax[r] - nm);
      rmax[r] = nm;
      const float p0 = __expf(a0 - nm);
      const float p1 = __expf(a1 - nm);
      float ts = p0 + p1;
      ts += __shfl_xor(ts, 1, 16);
      ts += __shfl_xor(ts, 2, 16);
      ts += __shfl_xor(ts, 4, 16);
      ts += __shfl_xor(ts, 8, 16);
      rsum[r] = rsum[r] * alpha + ts;
#pragma unroll
      for (int j = 0; j < 16; ++j) acc[j][r] *= alpha;
      const int row = r + hs*8;
      Psh[wave][row][lr]      = (_Float16)p0;
      Psh[wave][row][16 + lr] = (_Float16)p1;
    }

    // ---- reload P as A-fragment (row = lr; K = hs*16 + {0..15})
    v16h pa;
    {
      h8 lo = *(const h8*)&Psh[wave][lr][hs*16];
      h8 hi = *(const h8*)&Psh[wave][lr][hs*16 + 8];
#pragma unroll
      for (int i = 0; i < 8; ++i) { pa[i] = lo[i]; pa[8+i] = hi[i]; }
    }

    // ---- O += P * V  (16 dv tiles)
#pragma unroll
    for (int j = 0; j < 16; ++j) {
      v16h vb = ldb(&Vsh[cur][j*16 + lr][hs*16]);
      acc[j] = wmma_f16(pa, vb, acc[j]);
    }
  }

  // ---- normalize and store (f32 output)
  float inv[8];
#pragma unroll
  for (int r = 0; r < 8; ++r) inv[r] = 1.0f / rsum[r];
  float* Or = Out + ((size_t)b * SQ_ + q0) * DV_;
#pragma unroll
  for (int j = 0; j < 16; ++j)
#pragma unroll
    for (int r = 0; r < 8; ++r)
      Or[(size_t)(r + hs*8) * DV_ + j*16 + lr] = acc[j][r] * inv[r];
}

// ---------------------------------------------------------------------------
extern "C" void kernel_launch(void* const* d_in, const int* in_sizes, int n_in,
                              void* d_out, int out_size, void* d_ws, size_t ws_size,
                              hipStream_t stream) {
  const float* X  = (const float*)d_in[0];
  const float* Y  = (const float*)d_in[1];
  const float* Z  = (const float*)d_in[2];
  const float* Wk = (const float*)d_in[3];   // [D][D]
  const float* Wv = (const float*)d_in[4];   // [DV][D]
  // d_in[5] = rate (dropout is identity in eval) -> ignored
  float* Out = (float*)d_out;

  _Float16* Kh = (_Float16*)d_ws;            // [SK][D]  f16  (4 MB)
  _Float16* Vt = Kh + (size_t)SK_ * D_;      // [DV][SK] f16  (2 MB)

  // Kh[sk][d] = Y[sk] . Wk[d]
  {
    const int nT = SK_/16, tot = (D_/16) * nT;          // 8192 tiles
    proj_gemm16<<<dim3(tot/4), 128, 0, stream>>>(Wk, Y, Kh, nT, tot, D_);
  }
  // Vt[dv][sk] = Z[sk] . Wv[dv]
  {
    const int nT = DV_/16, tot = (SK_/16) * nT;         // 4096 tiles
    proj_gemm16<<<dim3(tot/4), 128, 0, stream>>>(Z, Wv, Vt, nT, tot, SK_);
  }
  attn_flash<<<dim3(SQ_/QB, B_), WAVES*32, 0, stream>>>(X, Kh, Vt, Out);
}
